// SparseAutoencoder_58385785422190
// MI455X (gfx1250) — matrix operand order repaired and checked by reading, not compiled, not run on hardware
//
#include <hip/hip_runtime.h>
#include <hip/hip_bf16.h>

// ---------------------------------------------------------------------------
// Sparse autoencoder forward for MI455X (gfx1250, wave32, WMMA).
//   recon [4096,1024], latent [4096,16384]  (concatenated in d_out)
// Main GEMM (137 GFLOP) on v_wmma_f32_16x16x32_f16, f32 accumulate.
// 4x4 WMMA tiles per wave -> 32 FLOP per L2 byte (operands are L2-resident:
// x_n f16 = 8 MB, w f16 = 32 MB, decoder = 64 MB; L2 = 192 MB).
// Decoder applied sparsely (32 cols/row) from a transposed copy.
// ---------------------------------------------------------------------------

#define BDIM 4096
#define HDIM 16384
#define DDIM 1024
#define TOPK 32

typedef __attribute__((ext_vector_type(8)))  _Float16 v8h;
typedef __attribute__((ext_vector_type(16))) _Float16 v16h;
typedef __attribute__((ext_vector_type(8)))  float    v8f;

// -------------------------- row L2-normalize -> f16 ------------------------
__global__ __launch_bounds__(256) void normalize_rows_f16(
    const float* __restrict__ src, _Float16* __restrict__ dst) {
  const int tid = threadIdx.x;
  const size_t rowOff = (size_t)blockIdx.x * DDIM;
  float4 val = *(const float4*)(src + rowOff + tid * 4);
  float ss = val.x * val.x + val.y * val.y + val.z * val.z + val.w * val.w;

  __shared__ float red[256];
  __shared__ float sScale;
  red[tid] = ss;
  __syncthreads();
  for (int s = 128; s > 0; s >>= 1) {
    if (tid < s) red[tid] += red[tid + s];
    __syncthreads();
  }
  if (tid == 0) sScale = 1.0f / fmaxf(sqrtf(red[0]), 1e-12f);
  __syncthreads();
  const float sc = sScale;
  _Float16* out = dst + rowOff + tid * 4;
  out[0] = (_Float16)(val.x * sc);
  out[1] = (_Float16)(val.y * sc);
  out[2] = (_Float16)(val.z * sc);
  out[3] = (_Float16)(val.w * sc);
}

// -------------------- decoder transpose [D,H] -> [H,D] ---------------------
__global__ __launch_bounds__(256) void transpose_dec(
    const float* __restrict__ in, float* __restrict__ out) {
  __shared__ float tile[32][33];
  const int tx = threadIdx.x;             // 0..31
  const int ty = threadIdx.y;             // 0..7
  const int hBase = blockIdx.x * 32;      // along H
  const int dBase = blockIdx.y * 32;      // along D
#pragma unroll
  for (int j = 0; j < 32; j += 8)
    tile[ty + j][tx] = in[(size_t)(dBase + ty + j) * HDIM + (hBase + tx)];
  __syncthreads();
#pragma unroll
  for (int j = 0; j < 32; j += 8)
    out[(size_t)(hBase + ty + j) * DDIM + (dBase + tx)] = tile[tx][ty + j];
}

// ------------- WMMA GEMM: acts[b,h] = f( x_n[b,:] . w[h,:] ) ---------------
// Block = 256 threads = 8 waves arranged 2(M) x 4(N); each wave owns a 64x64
// C macro-tile as 4x4 WMMA 16x16 tiles. Block covers 128(M) x 256(N).
// K loop: D=1024 in steps of 32. 16 WMMAs per 16 b128-loads per K step.
__global__ __launch_bounds__(256) void gemm_acts_wmma(
    const _Float16* __restrict__ xn,   // [B,D] row-major (K contiguous)
    const _Float16* __restrict__ wn,   // [H,D] row-major (K contiguous)
    float* __restrict__ acts) {        // [B,H]
  const int lane = threadIdx.x & 31;
  const int wave = threadIdx.x >> 5;
  const int l16  = lane & 15;
  const int hi   = lane >> 4;          // 0 or 1 (half-wave)
  const int m0 = blockIdx.y * 128 + (wave >> 2) * 64;
  const int n0 = blockIdx.x * 256 + (wave & 3) * 64;

  const _Float16* aBase[4];
  const _Float16* bBase[4];
#pragma unroll
  for (int i = 0; i < 4; ++i) {
    aBase[i] = xn + (size_t)(m0 + i * 16 + l16) * DDIM;
    bBase[i] = wn + (size_t)(n0 + i * 16 + l16) * DDIM;
  }

  v8f c[4][4];
#pragma unroll
  for (int mi = 0; mi < 4; ++mi)
#pragma unroll
    for (int ni = 0; ni < 4; ++ni) c[mi][ni] = (v8f){};

  for (int k0 = 0; k0 < DDIM; k0 += 32) {
    // A fragment (16x32 f16): lane half hi -> frag[0..15]=K(k0+hi*16..),
    // one contiguous 32B load per lane (same K order as B).
    v16h a[4];
#pragma unroll
    for (int mi = 0; mi < 4; ++mi) {
      a[mi] = *(const v16h*)(aBase[mi] + k0 + hi * 16);
    }
    // B fragment (32x16): lane l holds column n=l16, K = hi*16 .. hi*16+15,
    // i.e. one contiguous 32B run of w[n, :] per lane.
    v16h b[4];
#pragma unroll
    for (int ni = 0; ni < 4; ++ni)
      b[ni] = *(const v16h*)(bBase[ni] + k0 + hi * 16);

#pragma unroll
    for (int mi = 0; mi < 4; ++mi)
#pragma unroll
      for (int ni = 0; ni < 4; ++ni)
        c[mi][ni] = __builtin_amdgcn_wmma_f32_16x16x32_f16(
            false, a[mi], false, b[ni], (short)0, c[mi][ni], false, false);
  }

  // C/D layout: VGPR r -> (M = r + 8*hi, N = l16). Apply clip + activation.
#pragma unroll
  for (int mi = 0; mi < 4; ++mi) {
#pragma unroll
    for (int ni = 0; ni < 4; ++ni) {
#pragma unroll
      for (int r = 0; r < 8; ++r) {
        const int m = m0 + mi * 16 + hi * 8 + r;
        const int n = n0 + ni * 16 + l16;
        float cs = fminf(1.0f, fmaxf(-1.0f, c[mi][ni][r]));
        float act = 2.0f - sqrtf(fmaxf(0.0f, 2.0f - 2.0f * cs));
        acts[(size_t)m * HDIM + n] = act;
      }
    }
  }
}

// ---------- per-row top-32 radix select + sparse decode (recon) ------------
// One 256-thread block per row. acts are >= 0, so float order == u32 order.
__global__ __launch_bounds__(256) void topk_recon(
    float* __restrict__ latent,        // [B,H]: acts in, sparse latent out
    const float* __restrict__ decT,    // [H,D] transposed decoder
    const float* __restrict__ bias,    // [D]
    float* __restrict__ recon) {       // [B,D]
  const int tid = threadIdx.x;
  const int b = blockIdx.x;
  float* row = latent + (size_t)b * HDIM;

  unsigned v[64];                      // whole row lives in registers
#pragma unroll
  for (int i = 0; i < 64; ++i)
    v[i] = __float_as_uint(row[i * 256 + tid]);

  __shared__ unsigned hist[16];
  __shared__ unsigned sPrefix;
  __shared__ int sK;
  unsigned prefix = 0;
  int k = TOPK;                        // remaining rank among prefix-matches

  for (int p = 28; p >= 0; p -= 4) {
    if (tid < 16) hist[tid] = 0;
    __syncthreads();
    const unsigned maskHigh = (p == 28) ? 0u : ~((1u << (p + 4)) - 1u);
    unsigned long long h0 = 0, h1 = 0; // 16 packed 8-bit counters
#pragma unroll
    for (int i = 0; i < 64; ++i) {
      const unsigned x = v[i];
      if ((x & maskHigh) == prefix) {
        const unsigned nib = (x >> p) & 15u;
        const unsigned long long inc = 1ull << ((nib & 7u) * 8u);
        if (nib & 8u) h1 += inc; else h0 += inc;
      }
    }
#pragma unroll
    for (int j = 0; j < 8; ++j) {
      const unsigned c0 = (unsigned)((h0 >> (j * 8)) & 0xffull);
      const unsigned c1 = (unsigned)((h1 >> (j * 8)) & 0xffull);
      if (c0) atomicAdd(&hist[j], c0);
      if (c1) atomicAdd(&hist[j + 8], c1);
    }
    __syncthreads();
    if (tid == 0) {
      int cum = 0;
      unsigned chosen = 0;
      int kk = k;
      for (int bin = 15; bin >= 0; --bin) {
        const int c = (int)hist[bin];
        if (cum + c >= kk) { chosen = (unsigned)bin; sK = kk - cum; break; }
        cum += c;
      }
      sPrefix = prefix | (chosen << p);
    }
    __syncthreads();
    prefix = sPrefix;
    k = sK;
  }
  // prefix = bit pattern of the 32nd-largest value; k = #equal values to keep.

  __shared__ int   selIdx[TOPK];
  __shared__ float selVal[TOPK];
  __shared__ int nGt, nEq;
  if (tid == 0) { nGt = 0; nEq = 0; }
  __syncthreads();
  const int kEq = k;
  const int base = TOPK - kEq;         // #strictly-greater == TOPK - kEq
#pragma unroll
  for (int i = 0; i < 64; ++i) {
    const int e = i * 256 + tid;
    const unsigned x = v[i];
    const float f = __uint_as_float(x);
    float outv = 0.0f;
    if (x > prefix) {
      const int slot = atomicAdd(&nGt, 1);
      selIdx[slot] = e; selVal[slot] = f;
      outv = f;
    } else if (x == prefix) {
      const int r = atomicAdd(&nEq, 1);
      if (r < kEq) { selIdx[base + r] = e; selVal[base + r] = f; outv = f; }
    }
    row[e] = outv;                     // sparse latent written in place
  }
  __syncthreads();

  // recon[b,:] = bias + sum_k val_k * decT[idx_k,:]  (coalesced row gathers)
  float* rrow = recon + (size_t)b * DDIM;
#pragma unroll
  for (int j = 0; j < 4; ++j) {
    const int d = j * 256 + tid;
    float acc = bias[d];
#pragma unroll
    for (int t = 0; t < TOPK; ++t)
      acc += selVal[t] * decT[(size_t)selIdx[t] * DDIM + d];
    rrow[d] = acc;
  }
}

// ---------------------------------------------------------------------------
extern "C" void kernel_launch(void* const* d_in, const int* in_sizes, int n_in,
                              void* d_out, int out_size, void* d_ws, size_t ws_size,
                              hipStream_t stream) {
  const float* x    = (const float*)d_in[0];   // [4096,1024]
  const float* enc  = (const float*)d_in[1];   // [16384,1024]
  const float* decW = (const float*)d_in[2];   // [1024,16384]
  const float* bias = (const float*)d_in[3];   // [1024]

  char* ws = (char*)d_ws;
  _Float16* xn  = (_Float16*)ws;                               //  8 MB
  _Float16* wn  = (_Float16*)(ws + (size_t)8  * 1024 * 1024);  // 32 MB
  float*    decT = (float*)  (ws + (size_t)40 * 1024 * 1024);  // 64 MB

  float* recon  = (float*)d_out;                       // [4096,1024]
  float* latent = recon + (size_t)BDIM * DDIM;         // [4096,16384]

  normalize_rows_f16<<<BDIM, 256, 0, stream>>>(x, xn);
  normalize_rows_f16<<<HDIM, 256, 0, stream>>>(enc, wn);
  transpose_dec<<<dim3(HDIM / 32, DDIM / 32), dim3(32, 8), 0, stream>>>(decW, decT);
  gemm_acts_wmma<<<dim3(HDIM / 256, BDIM / 128), 256, 0, stream>>>(xn, wn, latent);
  topk_recon<<<BDIM, 256, 0, stream>>>(latent, decT, bias, recon);
}
